// ComplexSelfAttention_11089605558959
// MI455X (gfx1250) — matrix operand order repaired and checked by reading, compiled, not run on hardware
//
#include <hip/hip_runtime.h>
#include <stdint.h>

// ---------------------------------------------------------------------------
// Complex self-attention for MI455X (gfx1250), bf16 WMMA pipeline with
// double-buffered async (ASYNCcnt) LDS staging of the B operand.
// B=8, N=1024, D=512.  All GEMMs via V_WMMA_F32_16X16X32_BF16.
// ---------------------------------------------------------------------------

typedef __bf16 bf16_t;
typedef __attribute__((ext_vector_type(16))) __bf16       v16bf;
typedef __attribute__((ext_vector_type(8)))  __bf16       v8bf;
typedef __attribute__((ext_vector_type(8)))  float        v8f;
typedef __attribute__((ext_vector_type(8)))  unsigned int v8u;

#define D_DIM   512
#define N_SEQ   1024
#define B_BATCH 8
#define M_TOT   (B_BATCH * N_SEQ)   // 8192

// ---- WMMA fragment helpers (layouts per CDNA5 ISA 7.12.2, wave32) ----------

// A (16x32 bf16, row-major source, ld in elements):
// lanes 0-15: M=lane, elems 0..7 = K 0..7,  elems 8..15 = K 16..23
// lanes16-31: M=lane-16, elems 0..7 = K 8..15, elems 8..15 = K 24..31
__device__ __forceinline__ v16bf load_frag_a(const bf16_t* tile, int ld, int lane) {
    const int m  = lane & 15;
    const int ko = (lane & 16) ? 8 : 0;
    const bf16_t* r = tile + (size_t)m * ld + ko;
    v8bf lo = *(const v8bf*)(r);
    v8bf hi = *(const v8bf*)(r + 16);
    return __builtin_shufflevector(lo, hi, 0,1,2,3,4,5,6,7,8,9,10,11,12,13,14,15);
}

// Packed bf16 sign flip (BF16 WMMA has no A/B negate modifier; NEG = CNeg only)
__device__ __forceinline__ v16bf neg16(v16bf x) {
    v8u u = __builtin_bit_cast(v8u, x);
    u ^= 0x80008000u;
    return __builtin_bit_cast(v16bf, u);
}

__device__ __forceinline__ v8f wmma_bf16(v16bf a, v16bf b, v8f c) {
    return __builtin_amdgcn_wmma_f32_16x16x32_bf16(false, a, false, b,
                                                   (short)0, c, false, false);
}

// ---- CDNA5 async global->LDS copy (ASYNCcnt path) --------------------------

__device__ __forceinline__ void async_ld_b128(void* lds_dst, const void* gsrc) {
    // Generic LDS pointer: addr[31:0] is the DS byte offset (ISA 10.2).
    unsigned loff = (unsigned)(uintptr_t)lds_dst;
    asm volatile("global_load_async_to_lds_b128 %0, %1, off"
                 :: "v"(loff), "v"(gsrc) : "memory");
}

template <int N>
__device__ __forceinline__ void wait_async_le() {
#if __has_builtin(__builtin_amdgcn_s_wait_asynccnt)
    __builtin_amdgcn_s_wait_asynccnt((short)N);
#else
    if constexpr (N == 4) asm volatile("s_wait_asynccnt 4" ::: "memory");
    else                  asm volatile("s_wait_asynccnt 0" ::: "memory");
#endif
}

// Stage a pair of 128(N) x 32(K) bf16 tiles (Re/Im) into LDS.
// Source rows are contiguous in K with stride `ld`; g0/g1 point at
// (row 0 of the 128-row block, current kk).  256 threads x 2 chunks x 2 mats.
__device__ __forceinline__ void stage_pair(bf16_t* l0, bf16_t* l1,
                                           const bf16_t* g0, const bf16_t* g1,
                                           int ld, int tid) {
#pragma unroll
    for (int c = 0; c < 2; ++c) {
        const int chunk = tid + c * 256;       // 0..511
        const int nl  = chunk >> 2;            // row 0..127
        const int sub = (chunk & 3) * 8;       // 16-byte piece within 64B row
        async_ld_b128(l0 + nl * 32 + sub, g0 + (size_t)nl * ld + sub);
        async_ld_b128(l1 + nl * 32 + sub, g1 + (size_t)nl * ld + sub);
    }
}

// B fragment (32x16) read from a staged LDS tile: row nl holds K 0..31
// contiguously.  lanes 0-15: N=nl, K 0..7,16..23; lanes 16-31: K 8..15,24..31.
__device__ __forceinline__ v16bf load_frag_b_lds(const bf16_t* tile_row0,
                                                 int nl, int lane) {
    const int ko = (lane & 16) ? 8 : 0;
    const bf16_t* r = tile_row0 + (size_t)nl * 32 + ko;
    v8bf lo = *(const v8bf*)(r);
    v8bf hi = *(const v8bf*)(r + 16);
    return __builtin_shufflevector(lo, hi, 0,1,2,3,4,5,6,7,8,9,10,11,12,13,14,15);
}

// ---- fp32 -> bf16 cast (4-wide) --------------------------------------------

__global__ __launch_bounds__(256) void cast_f32_bf16(const float* __restrict__ in,
                                                     bf16_t* __restrict__ out, int n) {
    int i = (blockIdx.x * blockDim.x + threadIdx.x) * 4;
    if (i < n) {
        float4 f = *(const float4*)(in + i);
        out[i + 0] = (bf16_t)f.x;
        out[i + 1] = (bf16_t)f.y;
        out[i + 2] = (bf16_t)f.z;
        out[i + 3] = (bf16_t)f.w;
    }
}

// ---- complex linear: y = x @ W^T + b  (torch convention) -------------------
// yr = xr*Wr^T - xi*Wi^T + br ; yi = xr*Wi^T + xi*Wr^T + bi
// grid = (M_TOT/64, D/128), 256 threads = 8 waves (4 M-waves x 2 N-waves),
// wave tile 16(M) x 64(N).  Weight tiles double-buffered through LDS.
// TRANS_OUT: write y as (B, D, N) instead of (B*N, D) — used for v.

template <bool TRANS_OUT>
__global__ __launch_bounds__(256) void clinear_kernel(
    const bf16_t* __restrict__ xr, const bf16_t* __restrict__ xi,
    const bf16_t* __restrict__ Wr, const bf16_t* __restrict__ Wi,
    const float* __restrict__ br,  const float* __restrict__ bi,
    bf16_t* __restrict__ yr, bf16_t* __restrict__ yi)
{
    __shared__ __align__(64) bf16_t smw[2][2][128 * 32];   // 32 KB

    const int tid  = threadIdx.x;
    const int lane = tid & 31;
    const int wave = tid >> 5;
    const int m0   = blockIdx.x * 64 + (wave & 3) * 16;
    const int nblk = blockIdx.y * 128;
    const int nw   = (wave >> 2) * 64;      // wave's N offset inside block

    const bf16_t* Wr_blk = Wr + (size_t)nblk * D_DIM;
    const bf16_t* Wi_blk = Wi + (size_t)nblk * D_DIM;
    const bf16_t* xr_t   = xr + (size_t)m0 * D_DIM;
    const bf16_t* xi_t   = xi + (size_t)m0 * D_DIM;

    v8f accR[4] = {};
    v8f accI[4] = {};

    constexpr int T = D_DIM / 32;  // 16 K-steps
    stage_pair(smw[0][0], smw[0][1], Wr_blk,      Wi_blk,      D_DIM, tid);
    stage_pair(smw[1][0], smw[1][1], Wr_blk + 32, Wi_blk + 32, D_DIM, tid);

    for (int t = 0; t < T; ++t) {
        const int kk  = t * 32;
        const int cur = t & 1;
        if (t + 1 < T) wait_async_le<4>();
        else           wait_async_le<0>();
        __syncthreads();

        if (kk + 32 < D_DIM) __builtin_prefetch(xr_t + kk + 32, 0, 1);
        v16bf ar  = load_frag_a(xr_t + kk, D_DIM, lane);
        v16bf ai  = load_frag_a(xi_t + kk, D_DIM, lane);
        v16bf ain = neg16(ai);
#pragma unroll
        for (int q = 0; q < 4; ++q) {
            const int nl = nw + q * 16 + (lane & 15);
            v16bf bre = load_frag_b_lds(smw[cur][0], nl, lane);
            v16bf bim = load_frag_b_lds(smw[cur][1], nl, lane);
            accR[q] = wmma_bf16(ar,  bre, accR[q]);
            accR[q] = wmma_bf16(ain, bim, accR[q]);
            accI[q] = wmma_bf16(ar,  bim, accI[q]);
            accI[q] = wmma_bf16(ai,  bre, accI[q]);
        }
        __syncthreads();
        if (t + 2 < T)
            stage_pair(smw[cur][0], smw[cur][1],
                       Wr_blk + kk + 64, Wi_blk + kk + 64, D_DIM, tid);
    }

    const int mbase = m0 + ((lane & 16) ? 8 : 0);
#pragma unroll
    for (int q = 0; q < 4; ++q) {
        const int col = nblk + nw + q * 16 + (lane & 15);
        const float vbr = br[col];
        const float vbi = bi[col];
        if (TRANS_OUT) {
            // 8 consecutive N positions per lane -> one 16B packed store each
            const int b   = mbase >> 10;
            const int nl0 = mbase & (N_SEQ - 1);
            v8bf pr, pi;
#pragma unroll
            for (int j = 0; j < 8; ++j) {
                pr[j] = (bf16_t)(accR[q][j] + vbr);
                pi[j] = (bf16_t)(accI[q][j] + vbi);
            }
            const size_t o = (size_t)b * D_DIM * N_SEQ + (size_t)col * N_SEQ + nl0;
            *(v8bf*)(yr + o) = pr;
            *(v8bf*)(yi + o) = pi;
        } else {
#pragma unroll
            for (int j = 0; j < 8; ++j) {
                const size_t o = (size_t)(mbase + j) * D_DIM + col;
                yr[o] = (bf16_t)(accR[q][j] + vbr);
                yi[o] = (bf16_t)(accI[q][j] + vbi);
            }
        }
    }
}

// ---- complex scores + magnitude: |q k^T| / sqrt(D) -------------------------
// grid = (N/64, N/128, B).  K = D = 512.  Key tiles staged through LDS.

__global__ __launch_bounds__(256) void scores_abs_kernel(
    const bf16_t* __restrict__ qr, const bf16_t* __restrict__ qi,
    const bf16_t* __restrict__ kr, const bf16_t* __restrict__ ki,
    float* __restrict__ absS)
{
    __shared__ __align__(64) bf16_t smk[2][2][128 * 32];

    const int tid  = threadIdx.x;
    const int lane = tid & 31;
    const int wave = tid >> 5;
    const int b    = blockIdx.z;
    const int m0   = blockIdx.x * 64 + (wave & 3) * 16;
    const int nblk = blockIdx.y * 128;
    const int nw   = (wave >> 2) * 64;

    const bf16_t* qr_b = qr + (size_t)b * N_SEQ * D_DIM + (size_t)m0 * D_DIM;
    const bf16_t* qi_b = qi + (size_t)b * N_SEQ * D_DIM + (size_t)m0 * D_DIM;
    const bf16_t* kr_b = kr + (size_t)b * N_SEQ * D_DIM + (size_t)nblk * D_DIM;
    const bf16_t* ki_b = ki + (size_t)b * N_SEQ * D_DIM + (size_t)nblk * D_DIM;

    v8f accR[4] = {};
    v8f accI[4] = {};

    constexpr int T = D_DIM / 32;  // 16
    stage_pair(smk[0][0], smk[0][1], kr_b,      ki_b,      D_DIM, tid);
    stage_pair(smk[1][0], smk[1][1], kr_b + 32, ki_b + 32, D_DIM, tid);

    for (int t = 0; t < T; ++t) {
        const int kk  = t * 32;
        const int cur = t & 1;
        if (t + 1 < T) wait_async_le<4>();
        else           wait_async_le<0>();
        __syncthreads();

        v16bf ar  = load_frag_a(qr_b + kk, D_DIM, lane);
        v16bf ai  = load_frag_a(qi_b + kk, D_DIM, lane);
        v16bf ain = neg16(ai);
#pragma unroll
        for (int q = 0; q < 4; ++q) {
            const int nl = nw + q * 16 + (lane & 15);
            v16bf bre = load_frag_b_lds(smk[cur][0], nl, lane);
            v16bf bim = load_frag_b_lds(smk[cur][1], nl, lane);
            accR[q] = wmma_bf16(ar,  bre, accR[q]);  // qr.kr
            accR[q] = wmma_bf16(ain, bim, accR[q]);  // - qi.ki
            accI[q] = wmma_bf16(ar,  bim, accI[q]);  // qr.ki
            accI[q] = wmma_bf16(ai,  bre, accI[q]);  // + qi.kr
        }
        __syncthreads();
        if (t + 2 < T)
            stage_pair(smk[cur][0], smk[cur][1],
                       kr_b + kk + 64, ki_b + kk + 64, D_DIM, tid);
    }

    const float scale = 0.044194173824159216f;  // 1/sqrt(512)
    const int mbase = m0 + ((lane & 16) ? 8 : 0);
    float* out_b = absS + (size_t)b * N_SEQ * N_SEQ;
#pragma unroll
    for (int q = 0; q < 4; ++q) {
        const int col = nblk + nw + q * 16 + (lane & 15);
#pragma unroll
        for (int j = 0; j < 8; ++j) {
            const float sr = accR[q][j];
            const float si = accI[q][j];
            out_b[(size_t)(mbase + j) * N_SEQ + col] =
                sqrtf(sr * sr + si * si) * scale;
        }
    }
}

// ---- row softmax (1024 wide), fp32 in -> bf16 att out ----------------------

__global__ __launch_bounds__(256) void softmax_row_kernel(
    const float* __restrict__ s, bf16_t* __restrict__ att)
{
    __shared__ float red[8];
    __shared__ float bval;

    const size_t row = blockIdx.x;
    const float* p = s + row * N_SEQ;
    const int tid  = threadIdx.x;
    const int lane = tid & 31;
    const int wave = tid >> 5;

    float4 v = *(const float4*)(p + tid * 4);

    float mx = fmaxf(fmaxf(v.x, v.y), fmaxf(v.z, v.w));
#pragma unroll
    for (int off = 16; off >= 1; off >>= 1)
        mx = fmaxf(mx, __shfl_xor(mx, off, 32));
    if (lane == 0) red[wave] = mx;
    __syncthreads();
    if (tid == 0) {
        float m = red[0];
#pragma unroll
        for (int i = 1; i < 8; ++i) m = fmaxf(m, red[i]);
        bval = m;
    }
    __syncthreads();
    mx = bval;

    float4 e;
    e.x = __expf(v.x - mx);
    e.y = __expf(v.y - mx);
    e.z = __expf(v.z - mx);
    e.w = __expf(v.w - mx);
    float sum = e.x + e.y + e.z + e.w;
#pragma unroll
    for (int off = 16; off >= 1; off >>= 1)
        sum += __shfl_xor(sum, off, 32);
    __syncthreads();
    if (lane == 0) red[wave] = sum;
    __syncthreads();
    if (tid == 0) {
        float t = 0.f;
#pragma unroll
        for (int i = 0; i < 8; ++i) t += red[i];
        bval = t;
    }
    __syncthreads();
    const float inv = 1.0f / bval;

    bf16_t* o = att + row * N_SEQ + tid * 4;
    o[0] = (bf16_t)(e.x * inv);
    o[1] = (bf16_t)(e.y * inv);
    o[2] = (bf16_t)(e.z * inv);
    o[3] = (bf16_t)(e.w * inv);
}

// ---- att @ v (real and imag), fp32 out -------------------------------------
// grid = (N/64, D/128, B).  K = N = 1024.  v supplied transposed (B, D, N);
// value tiles staged through LDS.

__global__ __launch_bounds__(256) void attv_kernel(
    const bf16_t* __restrict__ att,
    const bf16_t* __restrict__ vTr, const bf16_t* __restrict__ vTi,
    float* __restrict__ outR, float* __restrict__ outI)
{
    __shared__ __align__(64) bf16_t smv[2][2][128 * 32];

    const int tid  = threadIdx.x;
    const int lane = tid & 31;
    const int wave = tid >> 5;
    const int b    = blockIdx.z;
    const int m0   = blockIdx.x * 64 + (wave & 3) * 16;
    const int nblk = blockIdx.y * 128;
    const int nw   = (wave >> 2) * 64;

    const bf16_t* att_b = att + (size_t)b * N_SEQ * N_SEQ + (size_t)m0 * N_SEQ;
    const bf16_t* vr_b  = vTr + (size_t)b * D_DIM * N_SEQ + (size_t)nblk * N_SEQ;
    const bf16_t* vi_b  = vTi + (size_t)b * D_DIM * N_SEQ + (size_t)nblk * N_SEQ;

    v8f accR[4] = {};
    v8f accI[4] = {};

    constexpr int T = N_SEQ / 32;  // 32
    stage_pair(smv[0][0], smv[0][1], vr_b,      vi_b,      N_SEQ, tid);
    stage_pair(smv[1][0], smv[1][1], vr_b + 32, vi_b + 32, N_SEQ, tid);

    for (int t = 0; t < T; ++t) {
        const int kk  = t * 32;
        const int cur = t & 1;
        if (t + 1 < T) wait_async_le<4>();
        else           wait_async_le<0>();
        __syncthreads();

        v16bf a = load_frag_a(att_b + kk, N_SEQ, lane);
#pragma unroll
        for (int q = 0; q < 4; ++q) {
            const int nl = nw + q * 16 + (lane & 15);
            v16bf bre = load_frag_b_lds(smv[cur][0], nl, lane);
            v16bf bim = load_frag_b_lds(smv[cur][1], nl, lane);
            accR[q] = wmma_bf16(a, bre, accR[q]);
            accI[q] = wmma_bf16(a, bim, accI[q]);
        }
        __syncthreads();
        if (t + 2 < T)
            stage_pair(smv[cur][0], smv[cur][1],
                       vr_b + kk + 64, vi_b + kk + 64, N_SEQ, tid);
    }

    const int mbase = m0 + ((lane & 16) ? 8 : 0);
#pragma unroll
    for (int q = 0; q < 4; ++q) {
        const int col = nblk + nw + q * 16 + (lane & 15);
#pragma unroll
        for (int j = 0; j < 8; ++j) {
            const size_t o = ((size_t)b * N_SEQ + mbase + j) * D_DIM + col;
            outR[o] = accR[q][j];
            outI[o] = accI[q][j];
        }
    }
}

// ---------------------------------------------------------------------------

extern "C" void kernel_launch(void* const* d_in, const int* in_sizes, int n_in,
                              void* d_out, int out_size, void* d_ws, size_t ws_size,
                              hipStream_t stream)
{
    (void)in_sizes; (void)n_in; (void)out_size; (void)ws_size;

    const float* x_re  = (const float*)d_in[0];
    const float* x_im  = (const float*)d_in[1];
    const float* Wq_re = (const float*)d_in[2];
    const float* Wq_im = (const float*)d_in[3];
    const float* bq_re = (const float*)d_in[4];
    const float* bq_im = (const float*)d_in[5];
    const float* Wk_re = (const float*)d_in[6];
    const float* Wk_im = (const float*)d_in[7];
    const float* bk_re = (const float*)d_in[8];
    const float* bk_im = (const float*)d_in[9];
    const float* Wv_re = (const float*)d_in[10];
    const float* Wv_im = (const float*)d_in[11];
    const float* bv_re = (const float*)d_in[12];
    const float* bv_im = (const float*)d_in[13];

    // ---- carve workspace ----
    char*  ws  = (char*)d_ws;
    size_t off = 0;
    auto carve = [&](size_t bytes) -> char* {
        char* p = ws + off;
        off = (off + bytes + 255) & ~(size_t)255;
        return p;
    };

    const size_t XB = (size_t)M_TOT * D_DIM * sizeof(bf16_t);     // 8 MiB
    const size_t WB = (size_t)D_DIM * D_DIM * sizeof(bf16_t);     // 512 KiB
    const size_t SB = (size_t)B_BATCH * N_SEQ * N_SEQ;            // 8M elems

    bf16_t* xr_b = (bf16_t*)carve(XB);
    bf16_t* xi_b = (bf16_t*)carve(XB);
    bf16_t* wqr  = (bf16_t*)carve(WB);
    bf16_t* wqi  = (bf16_t*)carve(WB);
    bf16_t* wkr  = (bf16_t*)carve(WB);
    bf16_t* wki  = (bf16_t*)carve(WB);
    bf16_t* wvr  = (bf16_t*)carve(WB);
    bf16_t* wvi  = (bf16_t*)carve(WB);
    bf16_t* q_r  = (bf16_t*)carve(XB);
    bf16_t* q_i  = (bf16_t*)carve(XB);
    bf16_t* k_r  = (bf16_t*)carve(XB);
    bf16_t* k_i  = (bf16_t*)carve(XB);
    bf16_t* vT_r = (bf16_t*)carve(XB);
    bf16_t* vT_i = (bf16_t*)carve(XB);
    float*  absS = (float*)carve(SB * sizeof(float));             // 32 MiB
    bf16_t* attb = (bf16_t*)carve(SB * sizeof(bf16_t));           // 16 MiB

    // ---- 1) fp32 -> bf16 casts ----
    {
        const int nx = M_TOT * D_DIM;       // 4,194,304
        const int nw = D_DIM * D_DIM;       // 262,144
        cast_f32_bf16<<<nx / 1024, 256, 0, stream>>>(x_re, xr_b, nx);
        cast_f32_bf16<<<nx / 1024, 256, 0, stream>>>(x_im, xi_b, nx);
        cast_f32_bf16<<<nw / 1024, 256, 0, stream>>>(Wq_re, wqr, nw);
        cast_f32_bf16<<<nw / 1024, 256, 0, stream>>>(Wq_im, wqi, nw);
        cast_f32_bf16<<<nw / 1024, 256, 0, stream>>>(Wk_re, wkr, nw);
        cast_f32_bf16<<<nw / 1024, 256, 0, stream>>>(Wk_im, wki, nw);
        cast_f32_bf16<<<nw / 1024, 256, 0, stream>>>(Wv_re, wvr, nw);
        cast_f32_bf16<<<nw / 1024, 256, 0, stream>>>(Wv_im, wvi, nw);
    }

    // ---- 2) complex projections (q, k row-major; v transposed) ----
    {
        dim3 grid(M_TOT / 64, D_DIM / 128);  // (128, 4)
        clinear_kernel<false><<<grid, 256, 0, stream>>>(xr_b, xi_b, wqr, wqi,
                                                        bq_re, bq_im, q_r, q_i);
        clinear_kernel<false><<<grid, 256, 0, stream>>>(xr_b, xi_b, wkr, wki,
                                                        bk_re, bk_im, k_r, k_i);
        clinear_kernel<true><<<grid, 256, 0, stream>>>(xr_b, xi_b, wvr, wvi,
                                                       bv_re, bv_im, vT_r, vT_i);
    }

    // ---- 3) |q k^T| * scale ----
    {
        dim3 grid(N_SEQ / 64, N_SEQ / 128, B_BATCH);  // (16, 8, 8)
        scores_abs_kernel<<<grid, 256, 0, stream>>>(q_r, q_i, k_r, k_i, absS);
    }

    // ---- 4) softmax rows -> bf16 att ----
    softmax_row_kernel<<<B_BATCH * N_SEQ, 256, 0, stream>>>(absS, attb);

    // ---- 5) att @ v -> [2, B, N, D] fp32 output ----
    {
        float* outR = (float*)d_out;
        float* outI = outR + (size_t)M_TOT * D_DIM;
        dim3 grid(N_SEQ / 64, D_DIM / 128, B_BATCH);  // (16, 4, 8)
        attv_kernel<<<grid, 256, 0, stream>>>(attb, vT_r, vT_i, outR, outI);
    }
}
